// GraphAttentionLayer_54812372632235
// MI455X (gfx1250) — matrix operand order repaired and checked by reading, not compiled, not run on hardware
//
#include <hip/hip_runtime.h>
#include <hip/hip_bf16.h>

#define B_    4
#define N_    2048
#define INF_  256
#define H_    8
#define D_    32
#define ALPHA 0.2f
#define LOG2E 1.44269504088896f
#define JB    128   // j-block staged in LDS for attention kernel

typedef __attribute__((ext_vector_type(16))) _Float16 v16h;
typedef __attribute__((ext_vector_type(8)))  _Float16 v8h;
typedef __attribute__((ext_vector_type(2)))  __fp16   v2fp;   // cvt_pkrtz result type
typedef __attribute__((ext_vector_type(8)))  float    v8f;

// ---------------------------------------------------------------------------
// Conversions: x -> f16; W -> f16 transposed to [h][d][k] so WMMA B-operand
// (16 consecutive K at fixed output column) is a contiguous v16h load.
// ---------------------------------------------------------------------------
__global__ void __launch_bounds__(256) k_cvt_x(const float* __restrict__ x,
                                               _Float16* __restrict__ xh, int n) {
  int idx = blockIdx.x * 256 + threadIdx.x;
  if (idx < n) xh[idx] = (_Float16)x[idx];
}

__global__ void __launch_bounds__(256) k_cvt_w(const float* __restrict__ W,
                                               _Float16* __restrict__ wt) {
  int o = blockIdx.x * 256 + threadIdx.x;  // enumerates h*D*K + d*K + k
  int k = o & (INF_ - 1);
  int rest = o >> 8;
  int d = rest & (D_ - 1);
  int h = rest >> 5;
  wt[o] = (_Float16)W[(h * INF_ + k) * D_ + d];
}

// ---------------------------------------------------------------------------
// Wh = x @ W per (b,h).  One wave computes a 16x32 tile (two 16x16 WMMA C
// tiles), K-loop over 256 in steps of 32.  Result stored TRANSPOSED as
// whT[bh][d][j] (f16) so the attention kernel's B loads are contiguous.
// ---------------------------------------------------------------------------
__global__ void __launch_bounds__(256) k_gemm_wh(const _Float16* __restrict__ xh,
                                                 const _Float16* __restrict__ wt,
                                                 _Float16* __restrict__ whT) {
  const int gw = (blockIdx.x * 256 + threadIdx.x) >> 5;  // 0..4095
  const int NT = N_ / 16;
  const int bh = gw / NT;
  const int it = gw % NT;
  const int b = bh / H_, h = bh % H_;
  const int lane = threadIdx.x & 31;
  const int hs = lane >> 4;        // lane-half select
  const int ll = lane & 15;        // A row / B-C column within tile
  const int i0 = it * 16;

  const _Float16* xrow = xh + (size_t)(b * N_ + i0 + ll) * INF_;
  const _Float16* w0 = wt + ((size_t)h * D_ + ll) * INF_;
  const _Float16* w1 = wt + ((size_t)h * D_ + 16 + ll) * INF_;

  v8f c0 = {}; v8f c1 = {};
#pragma unroll
  for (int k0 = 0; k0 < INF_; k0 += 32) {
    // A 16x32 f16 layout: lane half provides K runs [16*hs..+7] and [16*hs+8..+7] (contiguous)
    v8h alo = *(const v8h*)(xrow + k0 + 16 * hs);
    v8h ahi = *(const v8h*)(xrow + k0 + 16 * hs + 8);
    v16h a;
#pragma unroll
    for (int t = 0; t < 8; ++t) { a[t] = alo[t]; a[8 + t] = ahi[t]; }
    // B 32x16 f16 layout: lanes 0-15 K=0..15, lanes 16-31 K=16..31 (contiguous)
    v16h b0 = *(const v16h*)(w0 + k0 + 16 * hs);
    v16h b1 = *(const v16h*)(w1 + k0 + 16 * hs);
    c0 = __builtin_amdgcn_wmma_f32_16x16x32_f16(false, a, false, b0, (short)0, c0, false, false);
    c1 = __builtin_amdgcn_wmma_f32_16x16x32_f16(false, a, false, b1, (short)0, c1, false, false);
  }
  // C layout: (vgpr r, lane) -> row r+8*hs, col ll. Store transposed once.
  _Float16* o0 = whT + ((size_t)bh * D_ + ll) * N_;
  _Float16* o1 = whT + ((size_t)bh * D_ + 16 + ll) * N_;
#pragma unroll
  for (int r = 0; r < 8; ++r) {
    int orow = i0 + r + 8 * hs;
    o0[orow] = (_Float16)c0[r];
    o1[orow] = (_Float16)c1[r];
  }
}

// ---------------------------------------------------------------------------
// e_i / e_j: one wave per (bh, i) row; 32-lane dot product + shfl reduce.
// ---------------------------------------------------------------------------
__global__ void __launch_bounds__(256) k_e(const _Float16* __restrict__ whT,
                                           const float* __restrict__ a,
                                           float* __restrict__ ei,
                                           float* __restrict__ ej) {
  const int gw = (blockIdx.x * 256 + threadIdx.x) >> 5;  // 0..B*H*N-1
  const int bh = gw / N_;
  const int i  = gw % N_;
  const int h  = bh % H_;
  const int lane = threadIdx.x & 31;
  float wh = (float)whT[((size_t)bh * D_ + lane) * N_ + i];
  float pi = wh * a[h * 2 * D_ + lane];
  float pj = wh * a[h * 2 * D_ + D_ + lane];
#pragma unroll
  for (int off = 16; off > 0; off >>= 1) {
    pi += __shfl_xor(pi, off, 32);
    pj += __shfl_xor(pj, off, 32);
  }
  if (lane == 0) { ei[gw] = pi; ej[gw] = pj; }
}

// ---------------------------------------------------------------------------
// maxej[bh] = max_j e_j[bh][j] : 32 tiny block reductions.  Feeds the static
// per-row softmax shift (replaces an entire row-max pass over adj).
// ---------------------------------------------------------------------------
__global__ void __launch_bounds__(256) k_maxej(const float* __restrict__ ej,
                                               float* __restrict__ mxj) {
  __shared__ float red[8];
  const int bh = blockIdx.x;
  const int tid = threadIdx.x;
  const int wid = tid >> 5, lane = tid & 31;
  const float* p = ej + (size_t)bh * N_;
  float m = -3.0e38f;
#pragma unroll
  for (int k = 0; k < 8; ++k) m = fmaxf(m, p[tid + 256 * k]);
#pragma unroll
  for (int off = 16; off > 0; off >>= 1) m = fmaxf(m, __shfl_xor(m, off, 32));
  if (lane == 0) red[wid] = m;
  __syncthreads();
  if (tid == 0) {
    float t = red[0];
#pragma unroll
    for (int w = 1; w < 8; ++w) t = fmaxf(t, red[w]);
    mxj[bh] = t;
  }
}

// ---------------------------------------------------------------------------
// Fused single-pass attention:  h = softmax(mask(leaky(e_i+e_j))) @ Wh.
//   * static per-row shift = max(0, e_i + maxej)  =>  p = exp2((leaky-shift)L)
//     is in (0,1]: f16-safe, no online rescaling, exp computed ONCE per score.
//   * probabilities built directly in the WMMA A-operand register layout
//     (v_cvt_pk_rtz_f16_f32 packs two scores per op).
//   * row sums obtained by a 3rd WMMA against a ones-matrix: the sum lands in
//     the same lane/VGPR slot as the output rows -> per-lane normalize, no
//     cross-lane reduction at all.
//   * adj/e_j tiles staged in LDS (padded stride kills bank conflicts);
//     next adj tile prefetched (global_prefetch_b8) during compute.
// One block per (b, 16-row tile); one wave per head.
// ---------------------------------------------------------------------------
__global__ void __launch_bounds__(256) k_attn(const int* __restrict__ adj,
                                              const _Float16* __restrict__ whT,
                                              const float* __restrict__ ei,
                                              const float* __restrict__ ej,
                                              const float* __restrict__ mxj,
                                              float* __restrict__ out) {
  __shared__ int   adjt[16][JB + 4];
  __shared__ float ejt[H_][JB];
  const int NT = N_ / 16;
  const int b  = blockIdx.x / NT;
  const int it = blockIdx.x % NT;
  const int i0 = it * 16;
  const int tid = threadIdx.x;
  const int h = tid >> 5;          // wave == head
  const int lane = tid & 31;
  const int hs = lane >> 4, ll = lane & 15;
  const int bh = b * H_ + h;
  const int i  = i0 + ll;

  const float eiv   = ei[(size_t)bh * N_ + i];
  const float shift = fmaxf(0.0f, eiv + mxj[bh]);
  const float nsl   = -shift * LOG2E;          // folded into one fma per score
  const _Float16* w0 = whT + ((size_t)bh * D_ + ll) * N_;
  const _Float16* w1 = whT + ((size_t)bh * D_ + 16 + ll) * N_;
  const int*   adjbase = adj + (size_t)b * N_ * N_;
  const float* ejall   = ej + (size_t)b * H_ * N_;

  v16h ones;
#pragma unroll
  for (int t = 0; t < 16; ++t) ones[t] = (_Float16)1.0f;

  v8f c0 = {}; v8f c1 = {}; v8f cs = {};

  const int srr = tid >> 4;            // staging row / col for adj tile
  const int scc = (tid & 15) * 8;

  for (int jb = 0; jb < N_; jb += JB) {
    // Stage adj 16xJB tile (2048 ints, 8/thread, row-coalesced)
    {
      const int* src = adjbase + (size_t)(i0 + srr) * N_ + jb + scc;
#pragma unroll
      for (int t = 0; t < 8; ++t) adjt[srr][scc + t] = src[t];
      // Stage e_j H_xJB tile (1024 floats, 4/thread)
      int c2 = lane * 4;
      const float* esrc = ejall + (size_t)h * N_ + jb + c2;
#pragma unroll
      for (int t = 0; t < 4; ++t) ejt[h][c2 + t] = esrc[t];
    }
    __syncthreads();
    // Prefetch next j-block's adj tile into cache while we compute this one.
    if (jb + JB < N_) {
      __builtin_prefetch(adjbase + (size_t)(i0 + srr) * N_ + jb + JB + scc, 0, 1);
    }
#pragma unroll
    for (int k0 = 0; k0 < JB; k0 += 32) {
      // Build unnormalized probabilities directly in A 16x32 f16 layout
      v16h a;
      const int j1 = k0 + 16 * hs;
      const int j2 = k0 + 16 * hs + 8;
#pragma unroll
      for (int t = 0; t < 8; t += 2) {
        float p0, p1;
        {
          float s = eiv + ejt[h][j1 + t];
          s = fmaxf(s, ALPHA * s);                       // leakyReLU
          float e = exp2f(fmaf(s, LOG2E, nsl));          // exp(s - shift)
          p0 = adjt[ll][j1 + t] ? e : 0.0f;
        }
        {
          float s = eiv + ejt[h][j1 + t + 1];
          s = fmaxf(s, ALPHA * s);
          float e = exp2f(fmaf(s, LOG2E, nsl));
          p1 = adjt[ll][j1 + t + 1] ? e : 0.0f;
        }
        v2fp pk = __builtin_amdgcn_cvt_pkrtz(p0, p1);
        a[t] = (_Float16)pk[0]; a[t + 1] = (_Float16)pk[1];
        {
          float s = eiv + ejt[h][j2 + t];
          s = fmaxf(s, ALPHA * s);
          float e = exp2f(fmaf(s, LOG2E, nsl));
          p0 = adjt[ll][j2 + t] ? e : 0.0f;
        }
        {
          float s = eiv + ejt[h][j2 + t + 1];
          s = fmaxf(s, ALPHA * s);
          float e = exp2f(fmaf(s, LOG2E, nsl));
          p1 = adjt[ll][j2 + t + 1] ? e : 0.0f;
        }
        pk = __builtin_amdgcn_cvt_pkrtz(p0, p1);
        a[8 + t] = (_Float16)pk[0]; a[8 + t + 1] = (_Float16)pk[1];
      }
      const int kg = jb + k0 + 16 * hs;
      v16h b0 = *(const v16h*)(w0 + kg);
      v16h b1 = *(const v16h*)(w1 + kg);
      c0 = __builtin_amdgcn_wmma_f32_16x16x32_f16(false, a, false, b0,   (short)0, c0, false, false);
      c1 = __builtin_amdgcn_wmma_f32_16x16x32_f16(false, a, false, b1,   (short)0, c1, false, false);
      cs = __builtin_amdgcn_wmma_f32_16x16x32_f16(false, a, false, ones, (short)0, cs, false, false);
    }
    __syncthreads();
  }
  // Normalize per lane (row sum is resident in the matching C slot) and store
  // head-interleaved output: out[b][i][h*32 + d]
#pragma unroll
  for (int r = 0; r < 8; ++r) {
    float s = cs[r];
    float inv = s > 0.0f ? 1.0f / s : 0.0f;
    int orow = i0 + r + 8 * hs;
    float* op = out + ((size_t)b * N_ + orow) * (H_ * D_) + h * D_;
    op[ll]      = c0[r] * inv;
    op[16 + ll] = c1[r] * inv;
  }
}

// ---------------------------------------------------------------------------
extern "C" void kernel_launch(void* const* d_in, const int* in_sizes, int n_in,
                              void* d_out, int out_size, void* d_ws, size_t ws_size,
                              hipStream_t stream) {
  const float* x   = (const float*)d_in[0];
  const int*   adj = (const int*)d_in[1];
  const float* W   = (const float*)d_in[2];
  const float* a   = (const float*)d_in[3];
  float* out = (float*)d_out;

  char* ws = (char*)d_ws;
  size_t off = 0;
  auto alloc = [&](size_t bytes) {
    char* p = ws + off;
    off += (bytes + 255) & ~(size_t)255;
    return (void*)p;
  };
  _Float16* xh  = (_Float16*)alloc((size_t)B_ * N_ * INF_ * 2);     // 4 MB
  _Float16* wt  = (_Float16*)alloc((size_t)H_ * INF_ * D_ * 2);     // 128 KB
  _Float16* whT = (_Float16*)alloc((size_t)B_ * H_ * N_ * D_ * 2);  // 4 MB
  float*    ei  = (float*)alloc((size_t)B_ * H_ * N_ * 4);          // 256 KB
  float*    ej  = (float*)alloc((size_t)B_ * H_ * N_ * 4);          // 256 KB
  float*    mxj = (float*)alloc((size_t)B_ * H_ * 4);               // 128 B

  k_cvt_x<<<(B_ * N_ * INF_) / 256, 256, 0, stream>>>(x, xh, B_ * N_ * INF_);
  k_cvt_w<<<(H_ * INF_ * D_) / 256, 256, 0, stream>>>(W, wt);
  k_gemm_wh<<<(B_ * H_ * (N_ / 16)) / 8, 256, 0, stream>>>(xh, wt, whT);
  k_e<<<(B_ * H_ * N_) / 8, 256, 0, stream>>>(whT, a, ei, ej);
  k_maxej<<<B_ * H_, 256, 0, stream>>>(ej, mxj);
  k_attn<<<B_ * (N_ / 16), 256, 0, stream>>>(adj, whT, ei, ej, mxj, out);
}
